// E60CellPython_65189013618951
// MI455X (gfx1250) — matrix operand order repaired and checked by reading, not compiled, not run on hardware
//
#include <hip/hip_runtime.h>
#include <hip/hip_bf16.h>

#define TT 512
#define BB 64
#define DD 1024
#define BD (BB * DD)
#define LDS_STRIDE 1032   // 1024 + 8 bf16 pad -> 516 dwords/row -> bank step 4

typedef __bf16 bf16_t;
typedef __attribute__((ext_vector_type(16))) __bf16 v16bf;
typedef __attribute__((ext_vector_type(8)))  float  v8f;

// ---------- fragment loaders (wave32 WMMA 16x16x32 bf16 layouts) ----------
// A (16x32, MxK): lanes 0-15 hold row M=lane, K = {k..k+7, k+16..k+23};
//                 lanes 16-31 hold row M=lane-16, K = {k+8..k+15, k+24..k+31}.
// Caller bakes the +0/+8 half-wave element offset into p; fetch p[0..7], p[16..23].
__device__ __forceinline__ v16bf load_a_bf16(const bf16_t* p) {
    union { uint4 u[2]; v16bf v; } r;
    r.u[0] = *reinterpret_cast<const uint4*>(p);
    r.u[1] = *reinterpret_cast<const uint4*>(p + 16);
    return r.v;
}
// B (32x16, KxN, column-major == rows of W): lanes 0-15 hold K={0..7,16..23} of col
// N=lane; lanes 16-31 hold K={8..15,24..31}. Caller bakes +hi*8 into p; fetch p[0..7], p[16..23].
__device__ __forceinline__ v16bf load_b_bf16(const bf16_t* p) {
    union { uint4 u[2]; v16bf v; } r;
    r.u[0] = *reinterpret_cast<const uint4*>(p);
    r.u[1] = *reinterpret_cast<const uint4*>(p + 16);
    return r.v;
}

__device__ __forceinline__ v8f wmma_bf16(v16bf a, v16bf b, v8f c) {
    return __builtin_amdgcn_wmma_f32_16x16x32_bf16(
        /*neg_a=*/false, a, /*neg_b=*/false, b,
        /*c_mod=*/(short)0, c, /*reuse_a=*/false, /*reuse_b=*/false);
}

// pack 8 consecutive f32 -> uint4 of 8 bf16 (lowers to v_cvt_pk_bf16_f32 x4)
__device__ __forceinline__ uint4 pack8_bf16(const float* q) {
    float4 f0 = *reinterpret_cast<const float4*>(q);
    float4 f1 = *reinterpret_cast<const float4*>(q + 4);
    union { uint4 u; bf16_t h[8]; } pk;
    pk.h[0] = (__bf16)f0.x; pk.h[1] = (__bf16)f0.y;
    pk.h[2] = (__bf16)f0.z; pk.h[3] = (__bf16)f0.w;
    pk.h[4] = (__bf16)f1.x; pk.h[5] = (__bf16)f1.y;
    pk.h[6] = (__bf16)f1.z; pk.h[7] = (__bf16)f1.w;
    return pk.u;
}

// ---------- device-wide barrier (persistent kernel, all blocks resident) ----------
__device__ __forceinline__ void grid_barrier(unsigned* bar, unsigned target) {
    __syncthreads();
    if (threadIdx.x == 0) {
        __threadfence();  // release this block's global writes
        __hip_atomic_fetch_add(bar, 1u, __ATOMIC_RELEASE, __HIP_MEMORY_SCOPE_AGENT);
        while (__hip_atomic_load(bar, __ATOMIC_ACQUIRE, __HIP_MEMORY_SCOPE_AGENT) < target)
            __builtin_amdgcn_s_sleep(2);
    }
    __syncthreads();
    __threadfence();      // acquire: drop stale L0/L1 lines
}

// ---------- small prep kernels ----------
__global__ void cvt_f32_bf16_kernel(const float* __restrict__ src,
                                    bf16_t* __restrict__ dst, int n) {
    int i = blockIdx.x * blockDim.x + threadIdx.x;
    if (i < n) dst[i] = (__bf16)src[i];
}

__global__ void init_h_kernel(const float* __restrict__ h0,
                              float* __restrict__ h_all0,
                              bf16_t* __restrict__ hbf0, int n) {
    int i = blockIdx.x * blockDim.x + threadIdx.x;
    if (i < n) {
        float v = h0[i];
        h_all0[i] = v;
        hbf0[i]   = (__bf16)v;
    }
}

// ---------- phase 1: wx_all = x @ W_x^T + b  (stored into outs region) ----------
// Block = one 16-row M-block of x, 8 waves = 8 consecutive N-blocks.
// x rows are converted to bf16 once per block into LDS (shared by all 8 waves).
__global__ void __launch_bounds__(256) gemm_wx_kernel(
    const float* __restrict__ x,      // [T*B, D] row-major
    const bf16_t* __restrict__ Wxb,   // [D, D] row-major (bf16)
    const float* __restrict__ bias,   // [D]
    float* __restrict__ wx)           // [T*B, D]
{
    __shared__ __align__(16) bf16_t As[16 * LDS_STRIDE];   // ~33 KB

    const int tid  = threadIdx.x;
    const int wave = tid >> 5;
    const int lane = tid & 31;
    const int row  = lane & 15;
    const int hi   = lane >> 4;
    const int mb   = blockIdx.x >> 3;                 // 0..2047
    const int nb   = ((blockIdx.x & 7) << 3) + wave;  // 0..63
    const int m0 = mb * 16, n0 = nb * 16;

    // cooperative f32 -> bf16 conversion of the block's 16 x-rows into LDS
    for (int idx = tid; idx < 16 * 128; idx += 256) {
        const int r = idx >> 7;           // 0..15
        const int c = (idx & 127) * 8;    // element column, step 8
        *reinterpret_cast<uint4*>(&As[r * LDS_STRIDE + c]) =
            pack8_bf16(x + (size_t)(m0 + r) * DD + c);
    }
    __syncthreads();

    const float bv = bias[n0 + row];
    v8f acc;
#pragma unroll
    for (int r = 0; r < 8; ++r) acc[r] = bv;

    const bf16_t* albase = As  + row * LDS_STRIDE + (hi ? 8 : 0);       // LDS
    const bf16_t* bbase  = Wxb + (size_t)(n0 + row) * DD + hi * 8;      // global

#pragma unroll 4
    for (int k = 0; k < DD; k += 32) {
        v16bf a = load_a_bf16(albase + k);
        v16bf b = load_b_bf16(bbase + k);
        acc = wmma_bf16(a, b, acc);
    }

#pragma unroll
    for (int r = 0; r < 8; ++r)
        wx[(size_t)(m0 + r + hi * 8) * DD + n0 + row] = acc[r];
}

// ---------- phase 2: persistent sequential scan ----------
// grid = 32 blocks x 256 threads = 256 waves; wave w owns output tile
// (mb = w>>6, nb = w&63) of the 64x1024 state for every timestep.
// The block's 128 W_h rows (B operands, loop-invariant) are staged once in LDS
// (~258 KB of the WGP's 320 KB), so the steady-state k-loop is
// ds_load_b128 (B) + global_load_b128 (A, L2-resident h) + v_wmma.
__global__ void __launch_bounds__(256) rnn_scan_kernel(
    const bf16_t* __restrict__ Whb,      // [D, D] bf16
    bf16_t* __restrict__ hbf,            // [2][B*D] bf16 double buffer
    float* __restrict__ rowSq,           // [2][B]
    unsigned* __restrict__ bar,          // barrier counter (zeroed per launch)
    const float* __restrict__ log_alpha, // scalar
    float* __restrict__ outs,            // [T, B, D] (pre-filled with wx_all)
    float* __restrict__ h_all)           // [T+1, B, D]
{
    extern __shared__ __align__(16) char smem_raw[];
    bf16_t* Bs = reinterpret_cast<bf16_t*>(smem_raw);   // [128][LDS_STRIDE]

    const int tid  = threadIdx.x;
    const int wave = tid >> 5;
    const int lane = tid & 31;
    const int w    = blockIdx.x * 8 + wave;   // 0..255
    const int mb   = w >> 6;                  // 0..3
    const int row  = lane & 15;
    const int hi   = lane >> 4;
    const int m0   = mb * 16;
    const int nbase = (blockIdx.x & 7) * 128;          // first W_h row of block
    const int n0   = nbase + wave * 16;                // this wave's N-block
    const int gnc  = n0 + row;                         // C/D column of this lane
    const float alpha = __expf(log_alpha[0]);

    // stage the block's 128 W_h rows into LDS (once; loop-invariant)
    for (int idx = tid; idx < 128 * 128; idx += 256) {
        const int r = idx >> 7;            // 0..127
        const int c = (idx & 127) * 8;     // element column, step 8
        *reinterpret_cast<uint4*>(&Bs[r * LDS_STRIDE + c]) =
            *reinterpret_cast<const uint4*>(&Whb[(size_t)(nbase + r) * DD + c]);
    }
    __syncthreads();

    // h_prev tile carried in registers across all timesteps
    float hreg[8];
#pragma unroll
    for (int r = 0; r < 8; ++r)
        hreg[r] = h_all[(size_t)(m0 + r + hi * 8) * DD + gnc];

    const bf16_t* blbase = Bs + (wave * 16 + row) * LDS_STRIDE + hi * 8;   // LDS
    const size_t  a_off  = (size_t)(m0 + row) * DD + (hi ? 8 : 0);

    unsigned target = 0;
    for (int t = 0; t < TT; ++t) {
        const int buf = t & 1;

        // pre-zero the other rowSq buffer for step t+1 (its readers finished
        // before this step's compute phase; its writers start after barrier #2)
        if (blockIdx.x == 0 && tid < BB) rowSq[(buf ^ 1) * BB + tid] = 0.f;

        // C init = wx_t tile (staged in the outs region)
        const float* wxp = outs + (size_t)t * BD;
        v8f acc;
#pragma unroll
        for (int r = 0; r < 8; ++r)
            acc[r] = wxp[(size_t)(m0 + r + hi * 8) * DD + gnc];

        const bf16_t* abase = hbf + (size_t)buf * BD + a_off;
#pragma unroll 4
        for (int k = 0; k < DD; k += 32) {
            v16bf a = load_a_bf16(abase + k);     // global (L2-resident h)
            v16bf b = load_b_bf16(blbase + k);    // LDS
            acc = wmma_bf16(a, b, acc);
        }

        // h_raw = h_prev + alpha*tanh(pre_act); accumulate row sum-of-squares
        float praw[8];
#pragma unroll
        for (int r = 0; r < 8; ++r) {
            float pr = hreg[r] + alpha * tanhf(acc[r]);
            praw[r] = pr;
            float s = pr * pr;
            s += __shfl_xor(s, 1, 32);
            s += __shfl_xor(s, 2, 32);
            s += __shfl_xor(s, 4, 32);
            s += __shfl_xor(s, 8, 32);   // sum over the 16-lane half-wave
            if (row == 0) atomicAdd(&rowSq[buf * BB + m0 + r + hi * 8], s);
        }

        target += gridDim.x;
        grid_barrier(bar, target);

        // normalize, activations, write outputs + next-step bf16 operand
        float*  outp = outs  + (size_t)t * BD;
        float*  hap  = h_all + (size_t)(t + 1) * BD;
        bf16_t* hbp  = hbf   + (size_t)(buf ^ 1) * BD;
#pragma unroll
        for (int r = 0; r < 8; ++r) {
            const int gm = m0 + r + hi * 8;
            float ms = rowSq[buf * BB + gm] * (1.0f / DD);
            float hn = praw[r] * rsqrtf(ms + 1e-6f);
            hreg[r] = hn;
            float o = hn * hn / (1.0f + __expf(-hn));   // hn * silu(hn)
            size_t idx = (size_t)gm * DD + gnc;
            outp[idx] = o;
            hap[idx]  = hn;
            hbp[gm * DD + gnc] = (__bf16)hn;
        }

        target += gridDim.x;
        grid_barrier(bar, target);   // h_bf16[buf^1] published for step t+1
    }
}

// ---------- launcher ----------
extern "C" void kernel_launch(void* const* d_in, const int* in_sizes, int n_in,
                              void* d_out, int out_size, void* d_ws, size_t ws_size,
                              hipStream_t stream) {
    (void)in_sizes; (void)n_in; (void)out_size; (void)ws_size;
    const float* x         = (const float*)d_in[0];  // [T, B, D]
    const float* h0        = (const float*)d_in[1];  // [B, D]
    const float* W_h       = (const float*)d_in[2];  // [D, D]
    const float* W_x       = (const float*)d_in[3];  // [D, D]
    const float* bias      = (const float*)d_in[4];  // [D]
    const float* log_alpha = (const float*)d_in[5];  // scalar

    float* outs  = (float*)d_out;                    // [T*B*D]
    float* h_all = outs + (size_t)TT * BD;           // [(T+1)*B*D]

    char* ws = (char*)d_ws;
    bf16_t*   Whb   = (bf16_t*)ws;                                   // 2 MB
    bf16_t*   Wxb   = (bf16_t*)(ws + (size_t)DD * DD * 2);           // 2 MB
    bf16_t*   hbf   = (bf16_t*)(ws + (size_t)DD * DD * 4);           // 2 x 128 KB
    float*    rowSq = (float*)(ws + (size_t)DD * DD * 4 + 2 * BD * 2);
    unsigned* bar   = (unsigned*)(rowSq + 2 * BB);

    // reset barrier counter + rowSq every launch (graph-capturable)
    hipMemsetAsync(rowSq, 0, 2 * BB * sizeof(float) + sizeof(unsigned), stream);

    const int nW = DD * DD;
    cvt_f32_bf16_kernel<<<(nW + 255) / 256, 256, 0, stream>>>(W_h, Whb, nW);
    cvt_f32_bf16_kernel<<<(nW + 255) / 256, 256, 0, stream>>>(W_x, Wxb, nW);
    init_h_kernel<<<(BD + 255) / 256, 256, 0, stream>>>(h0, h_all, hbf, BD);

    // phase 1: wx_all -> outs region (2048 M-blocks x 8 N-groups)
    gemm_wx_kernel<<<2048 * 8, 256, 0, stream>>>(x, Wxb, bias, outs);

    // phase 2: persistent scan, 32 blocks (one per WGP: ~258 KB LDS each)
    const size_t scan_lds = (size_t)128 * LDS_STRIDE * sizeof(bf16_t);
    rnn_scan_kernel<<<32, 256, scan_lds, stream>>>(Whb, hbf, rowSq, bar,
                                                   log_alpha, outs, h_all);
}
